// AxisFuserLayer_40346922778712
// MI455X (gfx1250) — matrix operand order repaired and checked by reading, not compiled, not run on hardware
//
#include <hip/hip_runtime.h>
#include <hip/hip_bf16.h>
#include <math.h>

// ---------------------------------------------------------------------------
// Problem constants
// ---------------------------------------------------------------------------
#define BB        8
#define LL        1024
#define DMODEL    256
#define NHEAD     8
#define DHEAD     32            // 256/8
#define DINNER    512
#define DSTATE    16
#define DCONV     4
#define DTRANK    16
#define MROWS     (BB*LL)       // 8192
#define SROWS     (BB*3*LL)     // 24576  (attention flat rows)

typedef unsigned short ushort_t;
typedef __attribute__((ext_vector_type(16))) __bf16 v16bf;
typedef __attribute__((ext_vector_type(8)))  float  v8f;

union Frag16 { unsigned u[8]; v16bf v; };

__device__ __forceinline__ ushort_t bf16_1(float a) {
    unsigned u = __builtin_bit_cast(unsigned, a);
    u += 0x7FFFu + ((u >> 16) & 1u);            // round-to-nearest-even
    return (ushort_t)(u >> 16);
}

__device__ __forceinline__ float silu_f(float z) {
    return z / (1.f + __expf(-z));
}
__device__ __forceinline__ float softplus_f(float x) {
    return (x > 20.f) ? x : log1pf(__expf(x));
}

// ---------------------------------------------------------------------------
// fp32 -> bf16 weight conversion (done once per launch, outside the GEMM)
// ---------------------------------------------------------------------------
__global__ __launch_bounds__(256) void cvt_bf16_kernel(
    const float* __restrict__ src, ushort_t* __restrict__ dst, int n)
{
    int i = blockIdx.x * 256 + threadIdx.x;
    if (i < n) dst[i] = bf16_1(src[i]);
}

// ---------------------------------------------------------------------------
// WMMA GEMM:  C[M,N] = A[M,K] @ W[N,K]^T  (+bias if EPI==1)
// A, W pre-converted bf16 (ushort); fp32 accumulate/output.
// Block tile 128x128, 256 threads (8 waves in 4x2), wave tile 32x64
// -> 2x4 fragments -> 8 v_wmma per k-step per wave.
// M % 128 == 0 and K % 32 == 0 guaranteed; N guarded (N=48 case).
// LDS row stride 17 dwords: 17*r mod 32 distinct for r=0..15 (conflict-free).
// ---------------------------------------------------------------------------
template<int EPI>
__global__ __launch_bounds__(256) void wmma_gemm(
    const ushort_t* __restrict__ A, const ushort_t* __restrict__ W,
    const float* __restrict__ bias, float* __restrict__ C,
    int M, int N, int K)
{
    __shared__ unsigned As[128][17];   // 128 x 32 bf16 (16 dwords + pad)
    __shared__ unsigned Bs[128][17];

    const int m0   = blockIdx.x * 128;
    const int n0   = blockIdx.y * 128;
    const unsigned tid   = threadIdx.x;
    const unsigned lane  = tid & 31u;
    const unsigned wid   = tid >> 5;
    const unsigned wr    = wid >> 1;         // 0..3  (M strip of 32)
    const unsigned wc    = wid & 1u;         // 0..1  (N strip of 64)
    const unsigned r15   = lane & 15u;
    const unsigned khalf = lane >> 4;

    // staging role: each row of a tile handled by 2 threads (8 dwords each)
    const unsigned srow  = tid >> 1;         // 0..127
    const unsigned shalf = tid & 1u;         // 0..1
    const int      bn    = n0 + (int)srow;   // W row for B staging
    const bool     bok   = bn < N;

    v8f acc[2][4];
    #pragma unroll
    for (int i = 0; i < 2; ++i)
        #pragma unroll
        for (int j = 0; j < 4; ++j)
            acc[i][j] = v8f{0.f,0.f,0.f,0.f,0.f,0.f,0.f,0.f};

    for (int kt = 0; kt < K; kt += 32) {
        // ---- stage A tile: pure copy, uint4 (b128) global loads ----
        {
            const uint4* s4 = (const uint4*)(A + (size_t)(m0 + (int)srow) * K + kt) + shalf * 2;
            uint4 a0 = s4[0], a1 = s4[1];
            unsigned c0 = shalf * 8u;
            As[srow][c0 + 0] = a0.x; As[srow][c0 + 1] = a0.y;
            As[srow][c0 + 2] = a0.z; As[srow][c0 + 3] = a0.w;
            As[srow][c0 + 4] = a1.x; As[srow][c0 + 5] = a1.y;
            As[srow][c0 + 6] = a1.z; As[srow][c0 + 7] = a1.w;
            if (kt + 32 < K)
                __builtin_prefetch((const void*)((const char*)(s4) + 64), 0, 0);
        }
        // ---- stage W tile (zero-pad rows >= N) ----
        {
            uint4 a0 = uint4{0,0,0,0}, a1 = uint4{0,0,0,0};
            if (bok) {
                const uint4* s4 = (const uint4*)(W + (size_t)bn * K + kt) + shalf * 2;
                a0 = s4[0]; a1 = s4[1];
                if (kt + 32 < K)
                    __builtin_prefetch((const void*)((const char*)(s4) + 64), 0, 0);
            }
            unsigned c0 = shalf * 8u;
            Bs[srow][c0 + 0] = a0.x; Bs[srow][c0 + 1] = a0.y;
            Bs[srow][c0 + 2] = a0.z; Bs[srow][c0 + 3] = a0.w;
            Bs[srow][c0 + 4] = a1.x; Bs[srow][c0 + 5] = a1.y;
            Bs[srow][c0 + 6] = a1.z; Bs[srow][c0 + 7] = a1.w;
        }
        __syncthreads();

        // ---- fragments ----
        // A (16-bit 16x32, ISA 7.12.2): lane<16 -> K 0..7,16..23 ; lane>=16 -> 8..15,24..31
        Frag16 af[2], bf[4];
        #pragma unroll
        for (int ri = 0; ri < 2; ++ri) {
            unsigned row = wr * 32u + ri * 16u + r15;
            #pragma unroll
            for (int v = 0; v < 8; ++v)
                af[ri].u[v] = As[row][(unsigned)((v >> 2) * 8) + khalf * 4u + (unsigned)(v & 3)];
        }
        // B (32x16): same K order as A: lane<16 -> K 0..7,16..23 ; lane>=16 -> 8..15,24..31
        #pragma unroll
        for (int ci = 0; ci < 4; ++ci) {
            unsigned col = wc * 64u + ci * 16u + r15;
            #pragma unroll
            for (int v = 0; v < 8; ++v)
                bf[ci].u[v] = Bs[col][(unsigned)((v >> 2) * 8) + khalf * 4u + (unsigned)(v & 3)];
        }

        #pragma unroll
        for (int ri = 0; ri < 2; ++ri)
            #pragma unroll
            for (int ci = 0; ci < 4; ++ci)
                acc[ri][ci] = __builtin_amdgcn_wmma_f32_16x16x32_bf16(
                    false, af[ri].v, false, bf[ci].v,
                    (short)0, acc[ri][ci], false, false);

        __syncthreads();
    }

    // ---- store (C layout: VGPR r -> M = r + 8*(lane>>4), N = lane&15) ----
    #pragma unroll
    for (int ri = 0; ri < 2; ++ri) {
        #pragma unroll
        for (int ci = 0; ci < 4; ++ci) {
            int n = n0 + (int)(wc * 64u + ci * 16u + r15);
            if (n < N) {
                float bv = (EPI == 1) ? bias[n] : 0.f;
                int mbase = m0 + (int)(wr * 32u + ri * 16u + khalf * 8u);
                #pragma unroll
                for (int r = 0; r < 8; ++r)
                    C[(size_t)(mbase + r) * N + n] = acc[ri][ci][r] + bv;
            }
        }
    }
}

// ---------------------------------------------------------------------------
// Embeds: acc = accele @ acc_w^T + acc_b (bf16 out, feeds GEMM only);
//         ang likewise (fp32 out, needed later by LayerNorm).  K=12 (VALU).
// ---------------------------------------------------------------------------
__global__ __launch_bounds__(256) void embed_kernel(
    const float* __restrict__ accele, const float* __restrict__ angle,
    const float* __restrict__ acc_w, const float* __restrict__ acc_b,
    const float* __restrict__ ang_w, const float* __restrict__ ang_b,
    ushort_t* __restrict__ acc_e_bf, float* __restrict__ ang_e)
{
    int idx = blockIdx.x * 256 + threadIdx.x;       // MROWS*DMODEL
    int m = idx >> 8, n = idx & 255;
    const float* ar = accele + (size_t)m * 12;
    const float* gr = angle  + (size_t)m * 12;
    float s0 = acc_b[n], s1 = ang_b[n];
    #pragma unroll
    for (int j = 0; j < 12; ++j) {
        s0 += ar[j] * acc_w[n * 12 + j];
        s1 += gr[j] * ang_w[n * 12 + j];
    }
    acc_e_bf[idx] = bf16_1(s0);
    ang_e[idx] = s1;
}

// ---------------------------------------------------------------------------
// Depthwise causal conv (width 4) + SiLU on xi = xz[:, :512].
// Writes fp32 (for the scan) and bf16 (for the x_proj GEMM).
// ---------------------------------------------------------------------------
__global__ __launch_bounds__(256) void conv_silu_kernel(
    const float* __restrict__ xz, const float* __restrict__ conv_w,
    const float* __restrict__ conv_b, float* __restrict__ xc,
    ushort_t* __restrict__ xc_bf)
{
    int idx = blockIdx.x * 256 + threadIdx.x;       // MROWS*DINNER
    int m = idx >> 9, d = idx & 511;
    int l = m & (LL - 1);
    float s = conv_b[d];
    #pragma unroll
    for (int j = 0; j < DCONV; ++j) {
        int lj = l + j - (DCONV - 1);
        if (lj >= 0)
            s += xz[(size_t)(m - l + lj) * 1024 + d] * conv_w[d * DCONV + j];
    }
    float v = silu_f(s);
    xc[idx] = v;
    xc_bf[idx] = bf16_1(v);
}

// ---------------------------------------------------------------------------
// delta = softplus(dt @ dt_w^T + dt_b)   (K=16, VALU)
// ---------------------------------------------------------------------------
__global__ __launch_bounds__(256) void dtproj_kernel(
    const float* __restrict__ dbl, const float* __restrict__ dt_w,
    const float* __restrict__ dt_b, float* __restrict__ delta)
{
    int idx = blockIdx.x * 256 + threadIdx.x;       // MROWS*DINNER
    int m = idx >> 9, d = idx & 511;
    const float* dr = dbl + (size_t)m * 48;
    float s = dt_b[d];
    #pragma unroll
    for (int r = 0; r < DTRANK; ++r)
        s += dr[r] * dt_w[d * DTRANK + r];
    delta[idx] = softplus_f(s);
}

// ---------------------------------------------------------------------------
// Selective scan + gating.  Block per batch, lane per channel (512 lanes =
// 16 wave32).  Per-step B/C vectors broadcast via double-buffered LDS.
// Writes y_gated = (scan_y + u*Dp) * silu(z) directly as bf16 (GEMM input).
// ---------------------------------------------------------------------------
__global__ __launch_bounds__(512) void scan_kernel(
    const float* __restrict__ dbl, const float* __restrict__ delta,
    const float* __restrict__ xc, const float* __restrict__ xz,
    const float* __restrict__ A_log, const float* __restrict__ Dp,
    ushort_t* __restrict__ yg_bf)
{
    const int b = blockIdx.x;
    const int d = threadIdx.x;
    __shared__ float Bsh[2][DSTATE];
    __shared__ float Csh[2][DSTATE];

    float a[DSTATE], h[DSTATE];
    #pragma unroll
    for (int s = 0; s < DSTATE; ++s) {
        a[s] = -__expf(A_log[d * DSTATE + s]);
        h[s] = 0.f;
    }
    const float dpv = Dp[d];

    for (int l = 0; l < LL; ++l) {
        const int m  = b * LL + l;
        const int bf = l & 1;
        if (d < DSTATE)            Bsh[bf][d]          = dbl[(size_t)m * 48 + 16 + d];
        else if (d < 2 * DSTATE)   Csh[bf][d - DSTATE] = dbl[(size_t)m * 48 + 32 + (d - DSTATE)];
        __syncthreads();

        const float dl = delta[(size_t)m * DINNER + d];
        const float u  = xc[(size_t)m * DINNER + d];
        const float du = dl * u;
        float y = 0.f;
        #pragma unroll
        for (int s = 0; s < DSTATE; ++s) {
            h[s] = h[s] * __expf(dl * a[s]) + du * Bsh[bf][s];
            y   += h[s] * Csh[bf][s];
        }
        const float z = xz[(size_t)m * 1024 + DINNER + d];
        yg_bf[(size_t)m * DINNER + d] = bf16_1((y + u * dpv) * silu_f(z));
    }
}

// ---------------------------------------------------------------------------
// Three LayerNorms -> concatenated h (B, 3L, D) as bf16 (feeds attn_in GEMM).
// Wave per row (wave32), __shfl_xor reductions.
// ---------------------------------------------------------------------------
__global__ __launch_bounds__(256) void ln_concat_kernel(
    const float* __restrict__ x, const float* __restrict__ acc2,
    const float* __restrict__ ang_e,
    const float* __restrict__ nw,  const float* __restrict__ nb,
    const float* __restrict__ naw, const float* __restrict__ nab,
    const float* __restrict__ ngw, const float* __restrict__ ngb,
    ushort_t* __restrict__ hcat_bf)
{
    const int wid  = threadIdx.x >> 5;
    const int lane = threadIdx.x & 31;
    const int row  = blockIdx.x * 8 + wid;          // 0..SROWS-1
    const int nloc = row % (3 * LL);
    const int bb   = row / (3 * LL);
    const int seg  = nloc >> 10;
    const int l    = nloc & (LL - 1);

    const float* src; const float* w; const float* bs;
    if (seg == 0)      { src = x;     w = nw;  bs = nb;  }
    else if (seg == 1) { src = acc2;  w = naw; bs = nab; }
    else               { src = ang_e; w = ngw; bs = ngb; }

    const float* sp = src + (size_t)(bb * LL + l) * DMODEL;
    float v[8];
    float sum = 0.f;
    #pragma unroll
    for (int j = 0; j < 8; ++j) { v[j] = sp[j * 32 + lane]; sum += v[j]; }
    #pragma unroll
    for (int o = 16; o > 0; o >>= 1) sum += __shfl_xor(sum, o, 32);
    const float mean = sum * (1.f / DMODEL);
    float var = 0.f;
    #pragma unroll
    for (int j = 0; j < 8; ++j) { float t = v[j] - mean; var += t * t; }
    #pragma unroll
    for (int o = 16; o > 0; o >>= 1) var += __shfl_xor(var, o, 32);
    const float rstd = rsqrtf(var * (1.f / DMODEL) + 1e-5f);

    ushort_t* op = hcat_bf + (size_t)row * DMODEL;
    #pragma unroll
    for (int j = 0; j < 8; ++j) {
        int dd = j * 32 + lane;
        op[dd] = bf16_1((v[j] - mean) * rstd * w[dd] + bs[dd]);
    }
}

// ---------------------------------------------------------------------------
// Attention over the S=8 axis: per (n) block, 64 threads = (head, s) pairs.
// QKV slab (8 x 768) staged in LDS.  Output bf16 (feeds attn_out GEMM).
// ---------------------------------------------------------------------------
__global__ __launch_bounds__(64) void attn_kernel(
    const float* __restrict__ qkv, ushort_t* __restrict__ atto_bf)
{
    const int n = blockIdx.x;                       // 0..3071
    __shared__ float sq[8][768];
    for (int i = threadIdx.x; i < 8 * 768; i += 64) {
        int t = i / 768, c = i - t * 768;
        sq[t][c] = qkv[((size_t)t * (3 * LL) + n) * 768 + c];
    }
    __syncthreads();

    const int h = threadIdx.x >> 3;
    const int s = threadIdx.x & 7;

    float q[DHEAD];
    #pragma unroll
    for (int d = 0; d < DHEAD; ++d)
        q[d] = sq[s][h * DHEAD + d] * 0.17677669529663687f;   // 1/sqrt(32)

    float att[8]; float mx = -1e30f;
    #pragma unroll
    for (int t = 0; t < 8; ++t) {
        float dot = 0.f;
        #pragma unroll
        for (int d = 0; d < DHEAD; ++d)
            dot += q[d] * sq[t][DMODEL + h * DHEAD + d];
        att[t] = dot; mx = fmaxf(mx, dot);
    }
    float sum = 0.f;
    #pragma unroll
    for (int t = 0; t < 8; ++t) { att[t] = __expf(att[t] - mx); sum += att[t]; }
    const float inv = 1.f / sum;

    float o[DHEAD];
    #pragma unroll
    for (int d = 0; d < DHEAD; ++d) o[d] = 0.f;
    #pragma unroll
    for (int t = 0; t < 8; ++t) {
        const float w = att[t] * inv;
        #pragma unroll
        for (int d = 0; d < DHEAD; ++d)
            o[d] += w * sq[t][2 * DMODEL + h * DHEAD + d];
    }
    ushort_t* op = atto_bf + ((size_t)s * (3 * LL) + n) * DMODEL + h * DHEAD;
    #pragma unroll
    for (int d = 0; d < DHEAD; ++d) op[d] = bf16_1(o[d]);
}

// ---------------------------------------------------------------------------
// Final axis-2 re-concat: out[b, l, seg*256 + d] = hout[b, seg*1024 + l, d]
// ---------------------------------------------------------------------------
__global__ __launch_bounds__(256) void reorder_kernel(
    const float* __restrict__ hout, float* __restrict__ out)
{
    size_t idx = (size_t)blockIdx.x * 256 + threadIdx.x;    // SROWS*DMODEL
    int d    = (int)(idx & 255);
    size_t r = idx >> 8;
    int nloc = (int)(r % (3 * LL));
    int b    = (int)(r / (3 * LL));
    int seg  = nloc >> 10;
    int l    = nloc & (LL - 1);
    out[((size_t)b * LL + l) * (3 * DMODEL) + seg * DMODEL + d] = hout[idx];
}

// ---------------------------------------------------------------------------
// Launch
// ---------------------------------------------------------------------------
extern "C" void kernel_launch(void* const* d_in, const int* in_sizes, int n_in,
                              void* d_out, int out_size, void* d_ws, size_t ws_size,
                              hipStream_t stream)
{
    const float* x         = (const float*)d_in[0];
    const float* accele    = (const float*)d_in[1];
    const float* angle     = (const float*)d_in[2];
    const float* acc_w     = (const float*)d_in[3];
    const float* acc_b     = (const float*)d_in[4];
    const float* ang_w     = (const float*)d_in[5];
    const float* ang_b     = (const float*)d_in[6];
    const float* in_proj_w = (const float*)d_in[7];
    const float* conv_w    = (const float*)d_in[8];
    const float* conv_b    = (const float*)d_in[9];
    const float* x_proj_w  = (const float*)d_in[10];
    const float* dt_proj_w = (const float*)d_in[11];
    const float* dt_proj_b = (const float*)d_in[12];
    const float* A_log     = (const float*)d_in[13];
    const float* Dp        = (const float*)d_in[14];
    const float* out_proj_w= (const float*)d_in[15];
    const float* norm_w    = (const float*)d_in[16];
    const float* norm_b    = (const float*)d_in[17];
    const float* norm_acc_w= (const float*)d_in[18];
    const float* norm_acc_b= (const float*)d_in[19];
    const float* norm_ang_w= (const float*)d_in[20];
    const float* norm_ang_b= (const float*)d_in[21];
    const float* attn_in_w = (const float*)d_in[22];
    const float* attn_in_b = (const float*)d_in[23];
    const float* attn_out_w= (const float*)d_in[24];
    const float* attn_out_b= (const float*)d_in[25];
    float* out = (float*)d_out;

    // ---- workspace carve-up ----
    // fp32 region
    float* p = (float*)d_ws;
    float* ang_e = p;  p += (size_t)MROWS * DMODEL;     // 2M
    float* xz    = p;  p += (size_t)MROWS * 1024;       // 8M
    float* xc    = p;  p += (size_t)MROWS * DINNER;     // 4M
    float* dbl   = p;  p += (size_t)MROWS * 48;         // 0.4M
    float* delta = p;  p += (size_t)MROWS * DINNER;     // 4M
    float* acc2  = p;  p += (size_t)MROWS * DMODEL;     // 2M
    float* qkv   = p;  p += (size_t)SROWS * 768;        // 18.9M
    float* hout  = p;  p += (size_t)SROWS * DMODEL;     // 6.3M
    // bf16 region (ushort)
    ushort_t* q = (ushort_t*)p;
    ushort_t* acc_e_bf = q;  q += (size_t)MROWS * DMODEL;   // activations
    ushort_t* xc_bf    = q;  q += (size_t)MROWS * DINNER;
    ushort_t* yg_bf    = q;  q += (size_t)MROWS * DINNER;
    ushort_t* hcat_bf  = q;  q += (size_t)SROWS * DMODEL;
    ushort_t* atto_bf  = q;  q += (size_t)SROWS * DMODEL;
    ushort_t* w_in   = q;  q += (size_t)1024 * DMODEL;      // weights (bf16)
    ushort_t* w_xp   = q;  q += (size_t)48 * DINNER;
    ushort_t* w_op   = q;  q += (size_t)DMODEL * DINNER;
    ushort_t* w_ai   = q;  q += (size_t)768 * DMODEL;
    ushort_t* w_ao   = q;  q += (size_t)DMODEL * DMODEL;

    // 0. weight conversions (pure copies of small matrices; once per launch)
    cvt_bf16_kernel<<<(1024 * DMODEL) / 256, 256, 0, stream>>>(in_proj_w,  w_in, 1024 * DMODEL);
    cvt_bf16_kernel<<<(48 * DINNER) / 256, 256, 0, stream>>>(x_proj_w,   w_xp, 48 * DINNER);
    cvt_bf16_kernel<<<(DMODEL * DINNER) / 256, 256, 0, stream>>>(out_proj_w, w_op, DMODEL * DINNER);
    cvt_bf16_kernel<<<(768 * DMODEL) / 256, 256, 0, stream>>>(attn_in_w,  w_ai, 768 * DMODEL);
    cvt_bf16_kernel<<<(DMODEL * DMODEL) / 256, 256, 0, stream>>>(attn_out_w, w_ao, DMODEL * DMODEL);

    // 1. embeds
    embed_kernel<<<(MROWS * DMODEL) / 256, 256, 0, stream>>>(
        accele, angle, acc_w, acc_b, ang_w, ang_b, acc_e_bf, ang_e);

    // 2. in_proj: (8192x256)@(1024x256)^T -> xz
    wmma_gemm<0><<<dim3(MROWS / 128, 1024 / 128), 256, 0, stream>>>(
        acc_e_bf, w_in, nullptr, xz, MROWS, 1024, DMODEL);

    // 3. depthwise conv + SiLU
    conv_silu_kernel<<<(MROWS * DINNER) / 256, 256, 0, stream>>>(
        xz, conv_w, conv_b, xc, xc_bf);

    // 4. x_proj: (8192x512)@(48x512)^T -> dbl  (N=48 padded within one tile)
    wmma_gemm<0><<<dim3(MROWS / 128, 1), 256, 0, stream>>>(
        xc_bf, w_xp, nullptr, dbl, MROWS, 48, DINNER);

    // 5. dt projection + softplus
    dtproj_kernel<<<(MROWS * DINNER) / 256, 256, 0, stream>>>(
        dbl, dt_proj_w, dt_proj_b, delta);

    // 6. selective scan + gating (bf16 out)
    scan_kernel<<<BB, DINNER, 0, stream>>>(
        dbl, delta, xc, xz, A_log, Dp, yg_bf);

    // 7. out_proj: (8192x512)@(256x512)^T -> acc2
    wmma_gemm<0><<<dim3(MROWS / 128, 2), 256, 0, stream>>>(
        yg_bf, w_op, nullptr, acc2, MROWS, DMODEL, DINNER);

    // 8. three LayerNorms -> hcat_bf (B, 3L, D)
    ln_concat_kernel<<<SROWS / 8, 256, 0, stream>>>(
        x, acc2, ang_e, norm_w, norm_b, norm_acc_w, norm_acc_b,
        norm_ang_w, norm_ang_b, hcat_bf);

    // 9. attn_in: (24576x256)@(768x256)^T + b -> qkv
    wmma_gemm<1><<<dim3(SROWS / 128, 768 / 128), 256, 0, stream>>>(
        hcat_bf, w_ai, attn_in_b, qkv, SROWS, 768, DMODEL);

    // 10. tiny 8x8 attention over the batch axis (bf16 out)
    attn_kernel<<<3 * LL, 64, 0, stream>>>(qkv, atto_bf);

    // 11. attn_out: (24576x256)@(256x256)^T + b -> hout
    wmma_gemm<1><<<dim3(SROWS / 128, 2), 256, 0, stream>>>(
        atto_bf, w_ao, attn_out_b, hout, SROWS, DMODEL, DMODEL);

    // 12. final re-concat along feature axis
    reorder_kernel<<<(SROWS * DMODEL) / 256, 256, 0, stream>>>(hout, out);
}
